// Attention_37838661877873
// MI455X (gfx1250) — compile-verified, matrix operand order checked by reading, not run on hardware
//
#include <hip/hip_runtime.h>
#include <hip/hip_bf16.h>

// Sizes from the reference
#define B_ 64
#define S_ 2048
#define R_ 1024
#define H_ 512

typedef __attribute__((ext_vector_type(2)))  float    v2f;
typedef __attribute__((ext_vector_type(8)))  float    v8f;
typedef __attribute__((ext_vector_type(16))) _Float16 v16h;

// Hardware tanh (V_TANH_F32 trans op) when the builtin exists; else lib tanhf.
#if __has_builtin(__builtin_amdgcn_tanhf)
#define TANH_F32(x) __builtin_amdgcn_tanhf(x)
#else
#define TANH_F32(x) tanhf(x)
#endif

// ---------------------------------------------------------------------------
// Kernel 1: att_h[64,512] = h[64,1024] @ W[512,1024]^T + bias
// One wave (32 lanes) per 16x16 output tile, f32 WMMA over K in steps of 4.
// EXEC is all-1s (full 32-thread block, no divergence).
// ---------------------------------------------------------------------------
#if __has_builtin(__builtin_amdgcn_wmma_f32_16x16x4_f32)
__global__ __launch_bounds__(32) void h2att_wmma(const float* __restrict__ h,
                                                 const float* __restrict__ W,
                                                 const float* __restrict__ bias,
                                                 float* __restrict__ att_h) {
  const int n0   = blockIdx.x * 16;   // output col tile (H dim)
  const int m0   = blockIdx.y * 16;   // output row tile (B dim)
  const int lane = threadIdx.x;
  const int half = lane >> 4;         // 0: K pair {0,1}; 1: K pair {2,3}
  const int l15  = lane & 15;

  v8f c = {};
  const float* hrow = h + (size_t)(m0 + l15) * R_;  // A row (M = l15)
  const float* wrow = W + (size_t)(n0 + l15) * R_;  // B col = W row (B = W^T)

  for (int k = 0; k < R_; k += 4) {
    const int kb = k + 2 * half;
    v2f a, b;
    a.x = hrow[kb]; a.y = hrow[kb + 1];
    b.x = wrow[kb]; b.y = wrow[kb + 1];
    // (neg_a, A, neg_b, B, c_mod, C, reuse_a, reuse_b)
    c = __builtin_amdgcn_wmma_f32_16x16x4_f32(false, a, false, b,
                                              (short)0, c, false, false);
  }
  // D layout (16x16 f32): VGPR j -> row m0 + j + 8*half, col n0 + l15
  const int col = n0 + l15;
  const float bv = bias[col];
#pragma unroll
  for (int j = 0; j < 8; ++j)
    att_h[(size_t)(m0 + j + 8 * half) * H_ + col] = c[j] + bv;
}
#else
// Fallback on the codegen-confirmed f16 WMMA (K step 32), same tile mapping.
__global__ __launch_bounds__(32) void h2att_wmma(const float* __restrict__ h,
                                                 const float* __restrict__ W,
                                                 const float* __restrict__ bias,
                                                 float* __restrict__ att_h) {
  const int n0   = blockIdx.x * 16;
  const int m0   = blockIdx.y * 16;
  const int lane = threadIdx.x;
  const int half = lane >> 4;
  const int l15  = lane & 15;

  v8f c = {};
  const float* hrow = h + (size_t)(m0 + l15) * R_;
  const float* wrow = W + (size_t)(n0 + l15) * R_;

  for (int k = 0; k < R_; k += 32) {
    v16h a, b;
#pragma unroll
    for (int v = 0; v < 8; ++v) {
      const int kb = k + ((v & 4) ? 16 : 0) + 2 * (v & 3) + 8 * half;
      a[2 * v]     = (_Float16)hrow[kb];
      a[2 * v + 1] = (_Float16)hrow[kb + 1];
      b[2 * v]     = (_Float16)wrow[kb];
      b[2 * v + 1] = (_Float16)wrow[kb + 1];
    }
    c = __builtin_amdgcn_wmma_f32_16x16x32_f16(false, a, false, b,
                                               (short)0, c, false, false);
  }
  const int col = n0 + l15;
  const float bv = bias[col];
#pragma unroll
  for (int j = 0; j < 8; ++j)
    att_h[(size_t)(m0 + j + 8 * half) * H_ + col] = c[j] + bv;
}
#endif

// ---------------------------------------------------------------------------
// Kernel 2: scores[b,s] = sum_h tanh(p_att[b,s,h] + att_h[b,h]) * w_alpha[h] + b_alpha
// One wave per (b,s); 8 waves per block, all sharing the same b (S % 8 == 0).
// att_h[b,:] and w_alpha staged in LDS once per block (cuts vector-cache
// traffic 3x); p_att_feats streamed as coalesced float4 (b128). Hardware
// V_TANH keeps this pass memory-bound. Wave32 shuffle reduction.
// ---------------------------------------------------------------------------
__global__ __launch_bounds__(256) void scores_kernel(const float* __restrict__ p_att,
                                                     const float* __restrict__ att_h,
                                                     const float* __restrict__ w_alpha,
                                                     const float* __restrict__ b_alpha,
                                                     float* __restrict__ scores) {
  __shared__ float sh_ah[H_];
  __shared__ float sh_w[H_];
  const int tid  = threadIdx.x;
  const int wave = blockIdx.x * 8 + (tid >> 5);  // = b*S + s
  const int lane = tid & 31;
  const int b    = wave >> 11;                   // / 2048

  // Stage att_h[b,:] (512 f) and w_alpha (512 f) into LDS: 2 floats/thread.
  sh_ah[tid]       = att_h[(size_t)b * H_ + tid];
  sh_ah[tid + 256] = att_h[(size_t)b * H_ + tid + 256];
  sh_w[tid]        = w_alpha[tid];
  sh_w[tid + 256]  = w_alpha[tid + 256];
  __syncthreads();

  const float* p = p_att + (size_t)wave * H_;

  float acc = 0.f;
#pragma unroll
  for (int ch = 0; ch < 4; ++ch) {
    const int hb = ch * 128 + lane * 4;
    const float4 pv = *(const float4*)(p + hb);
    const float4 av = *(const float4*)(sh_ah + hb);   // ds_load_b128
    const float4 wv = *(const float4*)(sh_w + hb);    // ds_load_b128
    acc += TANH_F32(pv.x + av.x) * wv.x;
    acc += TANH_F32(pv.y + av.y) * wv.y;
    acc += TANH_F32(pv.z + av.z) * wv.z;
    acc += TANH_F32(pv.w + av.w) * wv.w;
  }
#pragma unroll
  for (int off = 16; off; off >>= 1) acc += __shfl_xor(acc, off, 32);
  if (lane == 0) scores[wave] = acc + b_alpha[0];
}

// ---------------------------------------------------------------------------
// Kernel 3: masked softmax + renormalize over S per batch row.
// softmax -> *mask -> /sum  collapses to  exp(s-max)*m / sum(exp(s-max)*m).
// One 256-thread block per b; 8 elements per thread.
// ---------------------------------------------------------------------------
__global__ __launch_bounds__(256) void softmax_kernel(const float* __restrict__ scores,
                                                      const int* __restrict__ mask,
                                                      float* __restrict__ weight) {
  __shared__ float red[8];
  const int b   = blockIdx.x;
  const int tid = threadIdx.x;
  const float* srow = scores + (size_t)b * S_;
  const int*   mrow = mask   + (size_t)b * S_;

  float v[8];
  int   m[8];
  float vmax = -__builtin_inff();
#pragma unroll
  for (int i = 0; i < 8; ++i) {
    const int s = tid + i * 256;
    v[i] = srow[s];
    m[i] = mrow[s];
    vmax = fmaxf(vmax, v[i]);
  }
#pragma unroll
  for (int off = 16; off; off >>= 1) vmax = fmaxf(vmax, __shfl_xor(vmax, off, 32));
  if ((tid & 31) == 0) red[tid >> 5] = vmax;
  __syncthreads();
  float bm = red[0];
#pragma unroll
  for (int w = 1; w < 8; ++w) bm = fmaxf(bm, red[w]);
  __syncthreads();

  float e[8];
  float sum = 0.f;
#pragma unroll
  for (int i = 0; i < 8; ++i) {
    e[i] = __expf(v[i] - bm) * (float)m[i];
    sum += e[i];
  }
#pragma unroll
  for (int off = 16; off; off >>= 1) sum += __shfl_xor(sum, off, 32);
  if ((tid & 31) == 0) red[tid >> 5] = sum;
  __syncthreads();
  float tot = 0.f;
#pragma unroll
  for (int w = 0; w < 8; ++w) tot += red[w];
  const float inv = 1.f / tot;
#pragma unroll
  for (int i = 0; i < 8; ++i)
    weight[(size_t)b * S_ + tid + i * 256] = e[i] * inv;
}

// ---------------------------------------------------------------------------
// Kernel 4: partial[chunk,b,:] = sum_{s in chunk} weight[b,s] * att_feats[b,s,:]
// The 512MB stream. Grid (64 b, 8 s-chunks) x 256 threads, float4 per thread
// covers all R=1024 per block -> 512 blocks * 8 waves of in-flight b128 loads.
// Weights staged in LDS; unroll 4 keeps >=4 b128 loads outstanding per thread;
// prefetch pulls the stream into cache ahead of use.
// ---------------------------------------------------------------------------
#define SCHUNK 256  // S_ / 8
__global__ __launch_bounds__(256) void attres_partial(const float* __restrict__ att_feats,
                                                      const float* __restrict__ weight,
                                                      float* __restrict__ partial) {
  __shared__ float wlds[SCHUNK];
  const int b     = blockIdx.x;
  const int chunk = blockIdx.y;
  const int tid   = threadIdx.x;
  const int s0    = chunk * SCHUNK;

  wlds[tid] = weight[(size_t)b * S_ + s0 + tid];
  __syncthreads();

  const float* F = att_feats + ((size_t)b * S_ + s0) * R_ + tid * 4;
  float4 acc = {0.f, 0.f, 0.f, 0.f};
#pragma unroll 4
  for (int s = 0; s < SCHUNK; ++s) {
    __builtin_prefetch(F + (size_t)(s + 8) * R_, 0, 0);  // global_prefetch_b8
    const float w  = wlds[s];
    const float4 f = *(const float4*)(F + (size_t)s * R_);
    acc.x += w * f.x;
    acc.y += w * f.y;
    acc.z += w * f.z;
    acc.w += w * f.w;
  }
  float4* outp = (float4*)(partial + ((size_t)chunk * B_ + b) * R_ + tid * 4);
  *outp = acc;
}

// ---------------------------------------------------------------------------
// Kernel 5: out[b,r] = sum_chunk partial[chunk,b,r]   (64*1024 outputs)
// ---------------------------------------------------------------------------
__global__ __launch_bounds__(256) void attres_reduce(const float* __restrict__ partial,
                                                     float* __restrict__ out) {
  const int idx = blockIdx.x * blockDim.x + threadIdx.x;  // 0 .. 65535
  float acc = 0.f;
#pragma unroll
  for (int c = 0; c < 8; ++c) acc += partial[(size_t)c * (B_ * R_) + idx];
  out[idx] = acc;
}

extern "C" void kernel_launch(void* const* d_in, const int* in_sizes, int n_in,
                              void* d_out, int out_size, void* d_ws, size_t ws_size,
                              hipStream_t stream) {
  const float* h         = (const float*)d_in[0];  // [64,1024]
  const float* att_feats = (const float*)d_in[1];  // [64,2048,1024]
  const float* p_att     = (const float*)d_in[2];  // [64,2048,512]
  const int*   masks     = (const int*)d_in[3];    // [64,2048]
  const float* W_h2att   = (const float*)d_in[4];  // [512,1024]
  const float* b_h2att   = (const float*)d_in[5];  // [512]
  const float* w_alpha   = (const float*)d_in[6];  // [512]
  const float* b_alpha   = (const float*)d_in[7];  // [1]
  float* out = (float*)d_out;                      // [64,1024]

  char* ws = (char*)d_ws;
  float* att_h_ws = (float*)(ws);                  //  64*512*4   = 128 KB
  float* scores   = (float*)(ws + 131072);         //  64*2048*4  = 512 KB
  float* weight   = (float*)(ws + 655360);         //  64*2048*4  = 512 KB
  float* partial  = (float*)(ws + 1179648);        //  8*64*1024*4= 2 MB

  hipLaunchKernelGGL(h2att_wmma, dim3(H_ / 16, B_ / 16), dim3(32), 0, stream,
                     h, W_h2att, b_h2att, att_h_ws);
  hipLaunchKernelGGL(scores_kernel, dim3((B_ * S_) / 8), dim3(256), 0, stream,
                     p_att, att_h_ws, w_alpha, b_alpha, scores);
  hipLaunchKernelGGL(softmax_kernel, dim3(B_), dim3(256), 0, stream,
                     scores, masks, weight);
  hipLaunchKernelGGL(attres_partial, dim3(B_, 8), dim3(256), 0, stream,
                     att_feats, weight, partial);
  hipLaunchKernelGGL(attres_reduce, dim3((B_ * R_) / 256), dim3(256), 0, stream,
                     partial, out);
}
